// PatchGCN_27032524161723
// MI455X (gfx1250) — matrix operand order repaired and checked by reading, not compiled, not run on hardware
//
#include <hip/hip_runtime.h>
#include <math.h>

// ---------------------------------------------------------------------------
// Types for CDNA5 WMMA (wave32): V_WMMA_F32_16X16X32_BF16
// ---------------------------------------------------------------------------
typedef __attribute__((ext_vector_type(16))) __bf16 v16bf;
typedef __attribute__((ext_vector_type(8)))  float  v8f;

union V16BF { unsigned short u[16]; uint4 q[2]; v16bf v; };

__device__ __forceinline__ unsigned short f2bf(float x) {
    unsigned u = __float_as_uint(x);
    unsigned r = u + 0x7fffu + ((u >> 16) & 1u);   // round-to-nearest-even
    return (unsigned short)(r >> 16);
}

// orderable encoding of float for atomicMax on unsigned
__device__ __forceinline__ unsigned fenc(float f) {
    unsigned u = __float_as_uint(f);
    return (u & 0x80000000u) ? ~u : (u | 0x80000000u);
}
__device__ __forceinline__ float fdec(unsigned e) {
    unsigned u = (e & 0x80000000u) ? (e & 0x7fffffffu) : ~e;
    return __uint_as_float(u);
}
#define ENC_NEG_INF 0x007FFFFFu   // fenc(-inf)

#define GEN_EPS 1e-7f
#define LN_EPS  1e-5f

// ---------------------------------------------------------------------------
// fp32 -> bf16 bulk convert (4-wide)
// ---------------------------------------------------------------------------
__global__ void k_cvt_bf16(const float* __restrict__ in,
                           unsigned short* __restrict__ out, size_t n)
{
    size_t i = ((size_t)blockIdx.x * blockDim.x + threadIdx.x) * 4;
    if (i + 4 <= n) {
        float4 v = *(const float4*)(in + i);
        unsigned short o0 = f2bf(v.x), o1 = f2bf(v.y), o2 = f2bf(v.z), o3 = f2bf(v.w);
        uint2 p;
        p.x = (unsigned)o0 | ((unsigned)o1 << 16);
        p.y = (unsigned)o2 | ((unsigned)o3 << 16);
        *(uint2*)(out + i) = p;
    } else {
        for (size_t j = i; j < n; ++j) out[j] = f2bf(in[j]);
    }
}

// Weight transpose+convert: Wt[n*K+k] = bf16( W[k*N+n] )   (row-major [K,N] in)
__global__ void k_wt_bf16(const float* __restrict__ W,
                          unsigned short* __restrict__ Wt, int K, int N)
{
    int idx = blockIdx.x * 256 + threadIdx.x;
    if (idx >= K * N) return;
    int n = idx / K, k = idx - n * K;
    Wt[idx] = f2bf(W[(size_t)k * N + n]);
}

// ---------------------------------------------------------------------------
// WMMA GEMM:  C[M,N] = epi( A_bf16[M,K](lda) @ Wt_bf16[N,K] + bias )
// EPI: 0 = bias, 1 = bias+relu, 2 = bias+tanh, 3 = C *= sigmoid(acc+bias)
// Block = 128 threads (4 waves); each wave computes a 32x32 C tile
// (4 accumulators). Per k-step: 8x global_load_b128 + 4x v_wmma. No LDS,
// no barriers -- B fragments are contiguous rows of the transposed bf16
// weights (tiny, resident in L2).
// Grid: x = ceil(M/128), y = N/32.
// ---------------------------------------------------------------------------
template <int EPI>
__device__ __forceinline__ void epi_tile(const v8f& acc, int mTile, int nTile,
                                         int lane, const float* bias,
                                         float* C, int ldc, int M)
{
    const int nloc = lane & 15;
    const int ncol = nTile + nloc;
    const float bv = bias ? bias[ncol] : 0.f;
    const int mbase = mTile + ((lane < 16) ? 0 : 8);
#pragma unroll
    for (int r = 0; r < 8; ++r) {
        int m = mbase + r;
        if (m < M) {
            float v = acc[r] + bv;
            size_t idx = (size_t)m * ldc + ncol;
            if      (EPI == 1) v = fmaxf(v, 0.f);
            else if (EPI == 2) v = tanhf(v);
            else if (EPI == 3) v = C[idx] * (1.f / (1.f + __expf(-v)));
            C[idx] = v;
        }
    }
}

template <int EPI>
__global__ __launch_bounds__(128)
void k_gemm(const unsigned short* __restrict__ A /*bf16 [M,K]*/, int lda,
            const unsigned short* __restrict__ Wt /*bf16 [N,K]*/,
            const float* __restrict__ bias,
            float* C, int ldc, int M, int N, int K)
{
    const int tid  = threadIdx.x;
    const int lane = tid & 31;
    const int wv   = tid >> 5;

    const int m0 = (blockIdx.x * 4 + wv) * 32;
    const int n0 = blockIdx.y * 32;

    v8f c00 = {}, c01 = {}, c10 = {}, c11 = {};

    int r0 = m0 + (lane & 15);      if (r0 >= M) r0 = M - 1;
    int r1 = m0 + 16 + (lane & 15); if (r1 >= M) r1 = M - 1;
    const int koffA  = (lane < 16) ? 0 : 8;     // 16-bit A 16x32 lane layout
    const int kbaseB = (lane < 16) ? 0 : 8;     // B frag: lanes 0-15 K=0..7,16..23
    const int nloc   = lane & 15;

    const unsigned short* A0 = A + (size_t)r0 * lda + koffA;
    const unsigned short* A1 = A + (size_t)r1 * lda + koffA;
    const unsigned short* B0 = Wt + (size_t)(n0 + nloc) * K + kbaseB;
    const unsigned short* B1 = Wt + (size_t)(n0 + 16 + nloc) * K + kbaseB;

    for (int k0 = 0; k0 < K; k0 += 32) {
        V16BF a0, a1, b0, b1;
        a0.q[0] = *(const uint4*)(A0 + k0);
        a0.q[1] = *(const uint4*)(A0 + k0 + 16);
        a1.q[0] = *(const uint4*)(A1 + k0);
        a1.q[1] = *(const uint4*)(A1 + k0 + 16);
        b0.q[0] = *(const uint4*)(B0 + k0);
        b0.q[1] = *(const uint4*)(B0 + k0 + 16);
        b1.q[0] = *(const uint4*)(B1 + k0);
        b1.q[1] = *(const uint4*)(B1 + k0 + 16);

        if (k0 + 32 < K) {
            __builtin_prefetch(A0 + k0 + 32, 0, 1);   // global_prefetch_b8
            __builtin_prefetch(A1 + k0 + 32, 0, 1);
        }

        c00 = __builtin_amdgcn_wmma_f32_16x16x32_bf16(false, a0.v, false, b0.v, (short)0, c00, false, false);
        c01 = __builtin_amdgcn_wmma_f32_16x16x32_bf16(false, a0.v, false, b1.v, (short)0, c01, false, false);
        c10 = __builtin_amdgcn_wmma_f32_16x16x32_bf16(false, a1.v, false, b0.v, (short)0, c10, false, false);
        c11 = __builtin_amdgcn_wmma_f32_16x16x32_bf16(false, a1.v, false, b1.v, (short)0, c11, false, false);
    }

    epi_tile<EPI>(c00, m0,      n0,      lane, bias, C, ldc, M);
    epi_tile<EPI>(c01, m0,      n0 + 16, lane, bias, C, ldc, M);
    epi_tile<EPI>(c10, m0 + 16, n0,      lane, bias, C, ldc, M);
    epi_tile<EPI>(c11, m0 + 16, n0 + 16, lane, bias, C, ldc, M);
}

// ---------------------------------------------------------------------------
// Edge softmax-aggregation kernels (GENConv message passing, D = 128)
// ---------------------------------------------------------------------------
__global__ void k_init_conv(unsigned* mEnc, float* den, float* num, int total)
{
    int i = blockIdx.x * blockDim.x + threadIdx.x;
    if (i < total) { mEnc[i] = ENC_NEG_INF; den[i] = 0.f; num[i] = 0.f; }
}

__global__ void k_edge_max(const float* __restrict__ h, int ldh,
                           const int* __restrict__ src, const int* __restrict__ dst,
                           const float* __restrict__ tp, unsigned* mEnc, int E)
{
    int e = blockIdx.x;
    if (e >= E) return;
    int f = threadIdx.x;                       // 0..127
    int s = src[e], d = dst[e];
    float t   = tp[0];
    float msg = fmaxf(h[(size_t)s * ldh + f], 0.f) + GEN_EPS;
    float sv  = msg * t;
    atomicMax(&mEnc[(size_t)d * 128 + f], fenc(sv));
}

__global__ void k_edge_acc(const float* __restrict__ h, int ldh,
                           const int* __restrict__ src, const int* __restrict__ dst,
                           const float* __restrict__ tp, const unsigned* __restrict__ mEnc,
                           float* den, float* num, int E)
{
    int e = blockIdx.x;
    if (e >= E) return;
    int f = threadIdx.x;
    int s = src[e], d = dst[e];
    float t   = tp[0];
    float msg = fmaxf(h[(size_t)s * ldh + f], 0.f) + GEN_EPS;
    float sv  = msg * t;
    float m   = fdec(mEnc[(size_t)d * 128 + f]);
    float ex  = __expf(sv - m);
    atomicAdd(&den[(size_t)d * 128 + f], ex);
    atomicAdd(&num[(size_t)d * 128 + f], msg * ex);
}

// aggbf[n,f] = bf16( (den>0 ? num/den : 0) + h[n,f] )   (GEMM1 A operand)
__global__ void k_agg(const float* __restrict__ h, int ldh,
                      const float* __restrict__ den, const float* __restrict__ num,
                      unsigned short* aggbf, int total /* M*128 */)
{
    int i = blockIdx.x * blockDim.x + threadIdx.x;
    if (i >= total) return;
    int n = i >> 7, f = i & 127;
    float d = den[i];
    float a = (d > 0.f) ? (num[i] / d) : 0.f;
    aggbf[i] = f2bf(a + h[(size_t)n * ldh + f]);
}

// ---------------------------------------------------------------------------
// LayerNorm + ReLU over rows of width 256; emits bf16 (GEMM2 A operand)
// ---------------------------------------------------------------------------
__global__ void k_ln_relu256(const float* __restrict__ x, const float* __restrict__ g,
                             const float* __restrict__ b, unsigned short* __restrict__ obf)
{
    __shared__ float red[256];
    int n = blockIdx.x, t = threadIdx.x;
    float v = x[(size_t)n * 256 + t];
    red[t] = v; __syncthreads();
    for (int s = 128; s > 0; s >>= 1) { if (t < s) red[t] += red[t + s]; __syncthreads(); }
    float mean = red[0] * (1.f / 256.f); __syncthreads();
    float d = v - mean;
    red[t] = d * d; __syncthreads();
    for (int s = 128; s > 0; s >>= 1) { if (t < s) red[t] += red[t + s]; __syncthreads(); }
    float var = red[0] * (1.f / 256.f);
    float y = d * rsqrtf(var + LN_EPS) * g[t] + b[t];
    obf[(size_t)n * 256 + t] = f2bf(fmaxf(y, 0.f));
}

// DeepGCN residual: feats[:,cur] = feats[:,prev] + relu(LN(c))   (width 128)
__global__ void k_res_ln(const float* __restrict__ c,
                         const float* __restrict__ g, const float* __restrict__ b,
                         const float* __restrict__ hprev, float* hdst)
{
    __shared__ float red[128];
    int n = blockIdx.x, t = threadIdx.x;
    float v = c[(size_t)n * 128 + t];
    red[t] = v; __syncthreads();
    for (int s = 64; s > 0; s >>= 1) { if (t < s) red[t] += red[t + s]; __syncthreads(); }
    float mean = red[0] * (1.f / 128.f); __syncthreads();
    float d = v - mean;
    red[t] = d * d; __syncthreads();
    for (int s = 64; s > 0; s >>= 1) { if (t < s) red[t] += red[t + s]; __syncthreads(); }
    float var = red[0] * (1.f / 128.f);
    float y = fmaxf(d * rsqrtf(var + LN_EPS) * g[t] + b[t], 0.f);
    hdst[(size_t)n * 512 + t] = hprev[(size_t)n * 512 + t] + y;
}

// ---------------------------------------------------------------------------
// Attention head
// ---------------------------------------------------------------------------
__global__ void k_score(const float* __restrict__ ag, const float* __restrict__ wc,
                        const float* __restrict__ bc, float* scores, int Nn)
{
    __shared__ float red[128];
    int n = blockIdx.x;
    if (n >= Nn) return;
    int t = threadIdx.x;
    float s = 0.f;
#pragma unroll
    for (int j = 0; j < 4; ++j) {
        int f = t + j * 128;
        s += ag[(size_t)n * 512 + f] * wc[f];
    }
    red[t] = s; __syncthreads();
    for (int sft = 64; sft > 0; sft >>= 1) { if (t < sft) red[t] += red[t + sft]; __syncthreads(); }
    if (t == 0) scores[n] = red[0] + bc[0];
}

__global__ void k_stats(const float* __restrict__ scores, int Nn, float* stats)
{
    __shared__ float red[512];
    int t = threadIdx.x;
    float mx = -3.4e38f;
    for (int i = t; i < Nn; i += 512) mx = fmaxf(mx, scores[i]);
    red[t] = mx; __syncthreads();
    for (int s = 256; s > 0; s >>= 1) { if (t < s) red[t] = fmaxf(red[t], red[t + s]); __syncthreads(); }
    float smax = red[0]; __syncthreads();
    float sum = 0.f;
    for (int i = t; i < Nn; i += 512) sum += __expf(scores[i] - smax);
    red[t] = sum; __syncthreads();
    for (int s = 256; s > 0; s >>= 1) { if (t < s) red[t] += red[t + s]; __syncthreads(); }
    if (t == 0) { stats[0] = smax; stats[1] = red[0]; }
}

__global__ void k_pool(const float* __restrict__ hp, const float* __restrict__ scores,
                       const float* __restrict__ stats, float* pooled, int Nn)
{
    __shared__ float red[256];
    int f = blockIdx.x, t = threadIdx.x;
    float smax = stats[0], inv = 1.f / stats[1];
    float s = 0.f;
    for (int n = t; n < Nn; n += 256)
        s += __expf(scores[n] - smax) * hp[(size_t)n * 512 + f];
    red[t] = s; __syncthreads();
    for (int sft = 128; sft > 0; sft >>= 1) { if (t < sft) red[t] += red[t + sft]; __syncthreads(); }
    if (t == 0) pooled[f] = red[0] * inv;
}

__global__ void k_head(const float* __restrict__ pooled, const float* __restrict__ hw,
                       const float* __restrict__ hb, float* out)
{
    int lane = threadIdx.x & 31;
    int c    = threadIdx.x >> 5;          // 4 waves -> 4 classes
    float s = 0.f;
    for (int f = lane; f < 512; f += 32) s += pooled[f] * hw[f * 4 + c];
    for (int o = 16; o > 0; o >>= 1) s += __shfl_down(s, o, 32);
    if (lane == 0) out[c] = s + hb[c];
}

// ---------------------------------------------------------------------------
// Host-side orchestration
// ---------------------------------------------------------------------------
extern "C" void kernel_launch(void* const* d_in, const int* in_sizes, int n_in,
                              void* d_out, int out_size, void* d_ws, size_t ws_size,
                              hipStream_t stream)
{
    (void)n_in; (void)out_size; (void)ws_size;

    const float* x      = (const float*)d_in[0];
    const int*   ei     = (const int*)  d_in[1];
    const float* fc_w   = (const float*)d_in[2];
    const float* fc_b   = (const float*)d_in[3];
    const float* cw1    = (const float*)d_in[4];
    const float* cb1    = (const float*)d_in[5];
    const float* cg     = (const float*)d_in[6];
    const float* cbe    = (const float*)d_in[7];
    const float* cw2    = (const float*)d_in[8];
    const float* cb2    = (const float*)d_in[9];
    const float* ct     = (const float*)d_in[10];
    const float* rg     = (const float*)d_in[11];
    const float* rb     = (const float*)d_in[12];
    const float* phi_w  = (const float*)d_in[13];
    const float* phi_b  = (const float*)d_in[14];
    const float* wa     = (const float*)d_in[15];
    const float* ba     = (const float*)d_in[16];
    const float* wb     = (const float*)d_in[17];
    const float* bb     = (const float*)d_in[18];
    const float* wc     = (const float*)d_in[19];
    const float* bc     = (const float*)d_in[20];
    const float* head_w = (const float*)d_in[21];
    const float* head_b = (const float*)d_in[22];

    const int Mn = in_sizes[0] / 1024;      // 50000 nodes
    const int E  = in_sizes[1] / 2;         // 600000 edges
    const int* srcI = ei;
    const int* dstI = ei + E;

    // ---- workspace layout (float units) ----
    float* ws      = (float*)d_ws;
    float* feats   = ws;                                   // [M,512] f32
    float* hp      = feats + (size_t)Mn * 512;             // [M,512] f32
    float* scratch = hp    + (size_t)Mn * 512;             // union region, M*1024 floats
    // phase A (fc):     xbf = bf16[M,1024]  (= M*512 float units)
    unsigned short* xbf = (unsigned short*)scratch;
    // phase B (convs):
    unsigned*       mEnc  = (unsigned*)scratch;                        // [M,128] u32
    float*          den   = scratch + (size_t)Mn * 128;                // [M,128]
    float*          num   = scratch + (size_t)Mn * 256;                // [M,128]
    float*          mid   = scratch + (size_t)Mn * 384;                // [M,256]
    float*          cbuf  = scratch + (size_t)Mn * 640;                // [M,128]
    unsigned short* aggbf = (unsigned short*)(scratch + (size_t)Mn * 768); // bf16[M,128]
    unsigned short* midbf = (unsigned short*)(scratch + (size_t)Mn * 832); // bf16[M,256]
    // phase C (head):
    float*          abuf    = scratch;                                     // [M,512] f32
    unsigned short* featsbf = (unsigned short*)(scratch + (size_t)Mn * 512); // bf16[M,512]
    unsigned short* hpbf    = (unsigned short*)(scratch + (size_t)Mn * 768); // bf16[M,512]
    float*          scores  = scratch + (size_t)Mn * 1024;             // [M]
    float*          stats   = scores + Mn;                             // [2]
    float*          pooled  = stats + 2;                               // [512]
    // persistent transposed bf16 weights (~2.2 MB)
    unsigned short* fcT  = (unsigned short*)(pooled + 512);            // [128,1024]
    unsigned short* w1T  = fcT  + 1024 * 128;                          // 3 x [256,128]
    unsigned short* w2T  = w1T  + 3 * 128 * 256;                       // 3 x [128,256]
    unsigned short* phiT = w2T  + 3 * 256 * 128;                       // [512,512]
    unsigned short* waT  = phiT + 512 * 512;                           // [512,512]
    unsigned short* wbT  = waT  + 512 * 512;                           // [512,512]

    const dim3 blk(128);
    const int  gx = (Mn + 127) / 128;

    // 0) transpose+convert all GEMM weights to bf16 [N,K]
    k_wt_bf16<<<(1024 * 128 + 255) / 256, 256, 0, stream>>>(fc_w, fcT, 1024, 128);
    for (int L = 0; L < 3; ++L) {
        k_wt_bf16<<<(128 * 256 + 255) / 256, 256, 0, stream>>>(
            cw1 + (size_t)L * 128 * 256, w1T + (size_t)L * 128 * 256, 128, 256);
        k_wt_bf16<<<(256 * 128 + 255) / 256, 256, 0, stream>>>(
            cw2 + (size_t)L * 256 * 128, w2T + (size_t)L * 256 * 128, 256, 128);
    }
    k_wt_bf16<<<(512 * 512 + 255) / 256, 256, 0, stream>>>(phi_w, phiT, 512, 512);
    k_wt_bf16<<<(512 * 512 + 255) / 256, 256, 0, stream>>>(wa, waT, 512, 512);
    k_wt_bf16<<<(512 * 512 + 255) / 256, 256, 0, stream>>>(wb, wbT, 512, 512);

    // 1) fc: xbf = bf16(x); feats[:,0:128] = relu(xbf @ fc_w + fc_b)
    {
        size_t nx = (size_t)Mn * 1024;
        k_cvt_bf16<<<(unsigned)((nx / 4 + 255) / 256), 256, 0, stream>>>(x, xbf, nx);
        k_gemm<1><<<dim3(gx, 4), blk, 0, stream>>>(xbf, 1024, fcT, fc_b,
                                                   feats, 512, Mn, 128, 1024);
    }

    // 2) three GENConv layers
    for (int L = 0; L < 3; ++L) {
        const float* hin = feats + L * 128;                // ld = 512
        int total = Mn * 128;
        k_init_conv<<<(total + 255) / 256, 256, 0, stream>>>(mEnc, den, num, total);
        k_edge_max<<<E, 128, 0, stream>>>(hin, 512, srcI, dstI, ct + L, mEnc, E);
        k_edge_acc<<<E, 128, 0, stream>>>(hin, 512, srcI, dstI, ct + L, mEnc, den, num, E);
        k_agg<<<(total + 255) / 256, 256, 0, stream>>>(hin, 512, den, num, aggbf, total);

        // MLP: mid = agg @ w1 + b1 ; LN ; relu -> bf16
        k_gemm<0><<<dim3(gx, 8), blk, 0, stream>>>(aggbf, 128,
                w1T + (size_t)L * 128 * 256, cb1 + L * 256, mid, 256, Mn, 256, 128);
        k_ln_relu256<<<Mn, 256, 0, stream>>>(mid, cg + L * 256, cbe + L * 256, midbf);

        if (L == 0) {
            // h1 = mid @ w2 + b2 -> feats[:,128:256]
            k_gemm<0><<<dim3(gx, 4), blk, 0, stream>>>(midbf, 256,
                    w2T + (size_t)L * 256 * 128, cb2 + L * 128,
                    feats + 128, 512, Mn, 128, 256);
        } else {
            // c = mid @ w2 + b2 ; h_{L+1} = h_L + relu(LN(c))
            k_gemm<0><<<dim3(gx, 4), blk, 0, stream>>>(midbf, 256,
                    w2T + (size_t)L * 256 * 128, cb2 + L * 128,
                    cbuf, 128, Mn, 128, 256);
            k_res_ln<<<Mn, 128, 0, stream>>>(cbuf, rg + (L - 1) * 128, rb + (L - 1) * 128,
                                             feats + L * 128, feats + (L + 1) * 128);
        }
    }

    // 3) path_phi: hp = relu(feats @ phi_w + phi_b)
    {
        size_t nf = (size_t)Mn * 512;
        k_cvt_bf16<<<(unsigned)((nf / 4 + 255) / 256), 256, 0, stream>>>(feats, featsbf, nf);
        k_gemm<1><<<dim3(gx, 16), blk, 0, stream>>>(featsbf, 512, phiT, phi_b,
                                                    hp, 512, Mn, 512, 512);
        k_cvt_bf16<<<(unsigned)((nf / 4 + 255) / 256), 256, 0, stream>>>(hp, hpbf, nf);
    }

    // 4) gated attention: abuf = tanh(hp@wa+ba);  abuf *= sigmoid(hp@wb+bb)
    k_gemm<2><<<dim3(gx, 16), blk, 0, stream>>>(hpbf, 512, waT, ba, abuf, 512, Mn, 512, 512);
    k_gemm<3><<<dim3(gx, 16), blk, 0, stream>>>(hpbf, 512, wbT, bb, abuf, 512, Mn, 512, 512);

    // 5) scores, softmax over nodes, weighted pool, classifier head
    k_score<<<Mn, 128, 0, stream>>>(abuf, wc, bc, scores, Mn);
    k_stats<<<1, 512, 0, stream>>>(scores, Mn, stats);
    k_pool<<<512, 256, 0, stream>>>(hp, scores, stats, pooled, Mn);
    k_head<<<1, 128, 0, stream>>>(pooled, head_w, head_b, (float*)d_out);
}
